// RNNGenerator_69114613730706
// MI455X (gfx1250) — matrix operand order repaired and checked by reading, not compiled, not run on hardware
//
#include <hip/hip_runtime.h>
#include <hip/hip_bf16.h>

// ---------------------------------------------------------------------------
// Fused 3-layer Elman RNN for MI455X (gfx1250, wave32, WMMA bf16).
//   h_t = tanh(x_t Wih^T + h_{t-1} Whh^T + (b_ih+b_hh))
// One block = 16 batch rows, 8 waves; wave w owns hidden N-tile w.
// Weights pre-quantized to padded bf16 in d_ws, preloaded as B-fragments in
// VGPRs; hidden states ping-pong in LDS (4 barriers / timestep); activations
// via native v_tanh_f32.
// ---------------------------------------------------------------------------

typedef __bf16 bf16_t;
typedef bf16_t v16bf __attribute__((ext_vector_type(16)));
typedef bf16_t v8bf  __attribute__((ext_vector_type(8)));
typedef float  v8f   __attribute__((ext_vector_type(8)));

union V16U { v16bf v; v8bf h[2]; };

// f32 -> bf16 bits, round-to-nearest-even
__device__ __forceinline__ unsigned short f2bf_bits(float f) {
    unsigned int u = __float_as_uint(f);
    unsigned int lsb = (u >> 16) & 1u;
    u += 0x7fffu + lsb;
    return (unsigned short)(u >> 16);
}

__device__ __forceinline__ unsigned int pack2bf(float a, float b) {
    return (unsigned int)f2bf_bits(a) | ((unsigned int)f2bf_bits(b) << 16);
}

__device__ __forceinline__ float fast_tanh(float x) {
#if __has_builtin(__builtin_amdgcn_tanhf)
    return __builtin_amdgcn_tanhf(x);          // v_tanh_f32 (CDNA5 TRANS op)
#elif __has_builtin(__builtin_amdgcn_tanh_f32)
    return __builtin_amdgcn_tanh_f32(x);
#else
    return tanhf(x);
#endif
}

// A-fragment (16x32 bf16, row-major source, row = M):
//   lanes 0-15 : K = {0..7} U {16..23};  lanes 16-31 : K = {8..15} U {24..31}
__device__ __forceinline__ v16bf load_a(const unsigned short* base, int row,
                                        int ld, int kt, int half) {
    const unsigned short* p = base + row * ld + kt * 32 + half * 8;
    V16U u;
    u.h[0] = *reinterpret_cast<const v8bf*>(p);
    u.h[1] = *reinterpret_cast<const v8bf*>(p + 16);
    return u.v;
}

// B-fragment (32x16 bf16). B[k][n] = W[n][k], so lane holds row n of W.
//   lanes 0-15 : K = {0..7} U {16..23};  lanes 16-31 : K = {8..15} U {24..31}
__device__ __forceinline__ v16bf load_b(const unsigned short* base, int row,
                                        int ld, int kt, int half) {
    const unsigned short* p = base + row * ld + kt * 32 + half * 8;
    V16U u;
    u.h[0] = *reinterpret_cast<const v8bf*>(p);
    u.h[1] = *reinterpret_cast<const v8bf*>(p + 16);
    return u.v;
}

__device__ __forceinline__ v8f wmma_bf16(v16bf a, v16bf b, v8f c) {
    return __builtin_amdgcn_wmma_f32_16x16x32_bf16(
        /*neg_a=*/false, a, /*neg_b=*/false, b,
        /*c_mod=*/(short)0, c, /*reuse_a=*/false, /*reuse_b=*/false);
}

// ---- weight scratch layout (bf16 elements, zero-padded row-major) ----------
#define OFF_WIH0 0        // [128][96]
#define OFF_WHH0 12288    // [128][128]
#define OFF_WIH1 28672    // [128][128]
#define OFF_WHH1 45056    // [128][128]
#define OFF_WIH2 61440    // [ 96][128]
#define OFF_WHH2 73728    // [ 96][96]
#define WQ_ELEMS 82944
#define BIAS_BYTE_OFF (WQ_ELEMS * 2)   // f32 [128 | 128 | 96]

__device__ __forceinline__ void fillw(unsigned short* dst, const float* src,
                                      int N, int K, int Npad, int Kpad,
                                      int tid0, int stride) {
    int tot = Npad * Kpad;
    for (int i = tid0; i < tot; i += stride) {
        int r = i / Kpad, k = i % Kpad;
        float v = (r < N && k < K) ? src[r * K + k] : 0.0f;
        dst[i] = f2bf_bits(v);
    }
}

__global__ __launch_bounds__(256) void rnn_prep_weights(
    const float* __restrict__ wih0, const float* __restrict__ whh0,
    const float* __restrict__ bih0, const float* __restrict__ bhh0,
    const float* __restrict__ wih1, const float* __restrict__ whh1,
    const float* __restrict__ bih1, const float* __restrict__ bhh1,
    const float* __restrict__ wih2, const float* __restrict__ whh2,
    const float* __restrict__ bih2, const float* __restrict__ bhh2,
    unsigned short* __restrict__ wq, float* __restrict__ bq) {
    const int stride = gridDim.x * blockDim.x;
    const int tid0 = blockIdx.x * blockDim.x + threadIdx.x;

    fillw(wq + OFF_WIH0, wih0, 128,  84, 128,  96, tid0, stride);
    fillw(wq + OFF_WHH0, whh0, 128, 128, 128, 128, tid0, stride);
    fillw(wq + OFF_WIH1, wih1, 128, 128, 128, 128, tid0, stride);
    fillw(wq + OFF_WHH1, whh1, 128, 128, 128, 128, tid0, stride);
    fillw(wq + OFF_WIH2, wih2,  84, 128,  96, 128, tid0, stride);
    fillw(wq + OFF_WHH2, whh2,  84,  84,  96,  96, tid0, stride);

    for (int i = tid0; i < 128; i += stride) bq[i]       = bih0[i] + bhh0[i];
    for (int i = tid0; i < 128; i += stride) bq[128 + i] = bih1[i] + bhh1[i];
    for (int i = tid0; i < 96;  i += stride)
        bq[256 + i] = (i < 84) ? (bih2[i] + bhh2[i]) : 0.0f;
}

__global__ __launch_bounds__(256) void rnn_fused(
    const float* __restrict__ x,            // (8192, 64, 84)
    const unsigned short* __restrict__ wq,  // padded bf16 weights
    const float* __restrict__ bq,           // combined biases
    float* __restrict__ out) {              // (8192, 64, 84)
    __shared__ __align__(16) unsigned short xs[16 * 96];       // staged x_t
    __shared__ __align__(16) unsigned short h0[2][16 * 128];   // ping-pong states
    __shared__ __align__(16) unsigned short h1[2][16 * 128];
    __shared__ __align__(16) unsigned short h2[2][16 * 96];

    const int tid  = threadIdx.x;
    const int w    = tid >> 5;       // wave id 0..7 (owns N-tile w)
    const int lane = tid & 31;
    const int l    = lane & 15;
    const int half = lane >> 4;
    const int b0   = blockIdx.x * 16;
    const int n    = w * 16 + l;     // this lane's output column

    // ---- preload all weight B-fragments into VGPRs (loop-invariant) ----
    const int n2 = (w < 6) ? n : 0;  // OOB guard for idle layer-2 waves
    v16bf Bih0[3], Bhh0[4], Bih1[4], Bhh1[4], Bih2[4], Bhh2[3];
#pragma unroll
    for (int kt = 0; kt < 3; ++kt) Bih0[kt] = load_b(wq + OFF_WIH0, n, 96, kt, half);
#pragma unroll
    for (int kt = 0; kt < 4; ++kt) Bhh0[kt] = load_b(wq + OFF_WHH0, n, 128, kt, half);
#pragma unroll
    for (int kt = 0; kt < 4; ++kt) Bih1[kt] = load_b(wq + OFF_WIH1, n, 128, kt, half);
#pragma unroll
    for (int kt = 0; kt < 4; ++kt) Bhh1[kt] = load_b(wq + OFF_WHH1, n, 128, kt, half);
#pragma unroll
    for (int kt = 0; kt < 4; ++kt) Bih2[kt] = load_b(wq + OFF_WIH2, n2, 128, kt, half);
#pragma unroll
    for (int kt = 0; kt < 3; ++kt) Bhh2[kt] = load_b(wq + OFF_WHH2, n2, 96, kt, half);

    const float bias0 = bq[n];
    const float bias1 = bq[128 + n];
    const float bias2 = (w < 6) ? bq[256 + n] : 0.0f;

    // ---- x staging plan: 16 rows x 42 float2 = 672 vector loads / timestep ----
    const float2* xp[3];
    int xoff[3];
    bool xval[3];
#pragma unroll
    for (int s = 0; s < 3; ++s) {
        int i = tid + 256 * s;
        xval[s] = (i < 672);
        int m = xval[s] ? (i / 42) : 0;
        int j = xval[s] ? (i - m * 42) : 0;
        xp[s] = reinterpret_cast<const float2*>(x + (size_t)(b0 + m) * 64 * 84) + j;
        xoff[s] = (m * 96 + 2 * j) >> 1;   // dword index into xs
    }
    // zero the constant padding columns (84..95) once
    for (int i = tid; i < 16 * 96; i += 256)
        if ((i % 96) >= 84) xs[i] = 0;

    // h_{-1} = 0 (both ping-pong buffers for safety)
    for (int i = tid; i < 2 * 16 * 128; i += 256) { (&h0[0][0])[i] = 0; (&h1[0][0])[i] = 0; }
    for (int i = tid; i < 2 * 16 * 96; i += 256) (&h2[0][0])[i] = 0;
    __syncthreads();

    for (int t = 0; t < 64; ++t) {
        const int rb = (t & 1) ^ 1;   // buffer holding h_{t-1}
        const int wb = t & 1;         // buffer receiving h_t

        // ---- stage x_t tile into LDS as bf16 ----
        unsigned int* xs32 = reinterpret_cast<unsigned int*>(xs);
#pragma unroll
        for (int s = 0; s < 3; ++s)
            if (xval[s]) {
                float2 v = xp[s][(size_t)t * 42];
                xs32[xoff[s]] = pack2bf(v.x, v.y);
            }
        __syncthreads();

        // ================= layer 0: 84(->96) -> 128 =================
        v8f c = {};
#pragma unroll
        for (int kt = 0; kt < 3; ++kt)
            c = wmma_bf16(load_a(xs, l, 96, kt, half), Bih0[kt], c);
#pragma unroll
        for (int kt = 0; kt < 4; ++kt)
            c = wmma_bf16(load_a(h0[rb], l, 128, kt, half), Bhh0[kt], c);
#pragma unroll
        for (int r = 0; r < 8; ++r)
            h0[wb][(r + half * 8) * 128 + n] = f2bf_bits(fast_tanh(c[r] + bias0));
        __syncthreads();

        // ================= layer 1: 128 -> 128 =================
        c = v8f{};
#pragma unroll
        for (int kt = 0; kt < 4; ++kt)
            c = wmma_bf16(load_a(h0[wb], l, 128, kt, half), Bih1[kt], c);
#pragma unroll
        for (int kt = 0; kt < 4; ++kt)
            c = wmma_bf16(load_a(h1[rb], l, 128, kt, half), Bhh1[kt], c);
#pragma unroll
        for (int r = 0; r < 8; ++r)
            h1[wb][(r + half * 8) * 128 + n] = f2bf_bits(fast_tanh(c[r] + bias1));
        __syncthreads();

        // ================= layer 2: 128 -> 84(->96) =================
        if (w < 6) {                    // wave-uniform: EXEC stays all-ones
            c = v8f{};
#pragma unroll
            for (int kt = 0; kt < 4; ++kt)
                c = wmma_bf16(load_a(h1[wb], l, 128, kt, half), Bih2[kt], c);
#pragma unroll
            for (int kt = 0; kt < 3; ++kt)
                c = wmma_bf16(load_a(h2[rb], l, 96, kt, half), Bhh2[kt], c);
#pragma unroll
            for (int r = 0; r < 8; ++r) {
                float v = fast_tanh(c[r] + bias2);
                int m = r + half * 8;
                h2[wb][m * 96 + n] = f2bf_bits(v);
                if (n < 84)
                    out[((size_t)(b0 + m) * 64 + t) * 84 + n] = v;
            }
        }
        __syncthreads();
    }
}

extern "C" void kernel_launch(void* const* d_in, const int* in_sizes, int n_in,
                              void* d_out, int out_size, void* d_ws, size_t ws_size,
                              hipStream_t stream) {
    (void)in_sizes; (void)n_in; (void)out_size;
    const float* x     = (const float*)d_in[0];
    const float* wih0  = (const float*)d_in[1];
    const float* whh0  = (const float*)d_in[2];
    const float* bih0  = (const float*)d_in[3];
    const float* bhh0  = (const float*)d_in[4];
    const float* wih1  = (const float*)d_in[5];
    const float* whh1  = (const float*)d_in[6];
    const float* bih1  = (const float*)d_in[7];
    const float* bhh1  = (const float*)d_in[8];
    const float* wih2  = (const float*)d_in[9];
    const float* whh2  = (const float*)d_in[10];
    const float* bih2  = (const float*)d_in[11];
    const float* bhh2  = (const float*)d_in[12];

    if (ws_size < (size_t)(BIAS_BYTE_OFF + 352 * sizeof(float))) return;
    unsigned short* wq = (unsigned short*)d_ws;
    float* bq = (float*)((char*)d_ws + BIAS_BYTE_OFF);

    rnn_prep_weights<<<64, 256, 0, stream>>>(wih0, whh0, bih0, bhh0,
                                             wih1, whh1, bih1, bhh1,
                                             wih2, whh2, bih2, bhh2, wq, bq);

    rnn_fused<<<8192 / 16, 256, 0, stream>>>(x, wq, bq, (float*)d_out);
}
